// Attention_21964462752059
// MI455X (gfx1250) — matrix operand order repaired and checked by reading, not compiled, not run on hardware
//
#include <hip/hip_runtime.h>
#include <hip/hip_bf16.h>
#include <math.h>

typedef _Float16 half_t;
typedef __attribute__((ext_vector_type(16))) _Float16 v16h;
typedef __attribute__((ext_vector_type(8)))  _Float16 v8h;
typedef __attribute__((ext_vector_type(8)))  float    v8f;

#define DIM      1024
#define NHEADS   16
#define HDIM     64
#define BATCH    2
#define SEQ      2048
#define TOK      (BATCH * SEQ)          // 4096
#define F3       (3 * DIM)              // 3072
#define SCALE    0.125f                 // 64^-0.5

__device__ __forceinline__ v8f wmma16(v16h a, v16h b, v8f c) {
    return __builtin_amdgcn_wmma_f32_16x16x32_f16(
        /*neg_a=*/false, a, /*neg_b=*/false, b,
        /*c_mod=*/(short)0, c, /*reuse_a=*/false, /*reuse_b=*/false);
}

// 16-bit A-matrix 16x32 fragment: row = (lane&15); per lane, halfs at
// k = kk + half16*16 + {0..15} (contiguous, matches B loads).
__device__ __forceinline__ v16h load_a_frag(const half_t* row, int kk, int half16) {
    const half_t* p = row + kk + half16 * 16;
    v8h lo = *(const v8h*)(p);
    v8h hi = *(const v8h*)(p + 8);
    return __builtin_shufflevector(lo, hi, 0,1,2,3,4,5,6,7,8,9,10,11,12,13,14,15);
}

// ---------------------------------------------------------------- convert
__global__ __launch_bounds__(256) void f32_to_f16_kernel(const float* __restrict__ src,
                                                         half_t* __restrict__ dst, int n) {
    int i = blockIdx.x * 256 + threadIdx.x;
    if (i < n) dst[i] = (half_t)src[i];
}

// ---------------------------------------------------------------- QKV GEMM
// 64(M) x 64(N) per wave: 16 WMMAs per k-step, 8 b128-load pairs.
// which/h/bb are wave-uniform (f0, m0 multiples of 64) -> scalar branch epilogue.
__global__ __launch_bounds__(128) void qkv_kernel(const half_t* __restrict__ xh,
                                                  const half_t* __restrict__ wh,
                                                  half_t* __restrict__ Qh,
                                                  half_t* __restrict__ Kh,
                                                  half_t* __restrict__ Vth) {
    const int lane   = threadIdx.x & 31;
    const int wv     = threadIdx.x >> 5;
    const int half16 = lane >> 4;
    const int ln     = lane & 15;
    const int m0     = blockIdx.x * 64;
    const int f0     = blockIdx.y * 256 + wv * 64;

    const half_t* arow[4];
#pragma unroll
    for (int rb = 0; rb < 4; ++rb)
        arow[rb] = xh + (size_t)(m0 + rb * 16 + ln) * DIM;
    const half_t* brow[4];
#pragma unroll
    for (int ct = 0; ct < 4; ++ct)
        brow[ct] = wh + (size_t)(f0 + ct * 16 + ln) * DIM + half16 * 16;

    v8f zero = {};
    v8f acc[4][4];
#pragma unroll
    for (int rb = 0; rb < 4; ++rb)
#pragma unroll
        for (int ct = 0; ct < 4; ++ct) acc[rb][ct] = zero;

    for (int kk = 0; kk < DIM; kk += 32) {
        v16h a[4], b[4];
#pragma unroll
        for (int rb = 0; rb < 4; ++rb) {
            a[rb] = load_a_frag(arow[rb], kk, half16);
            __builtin_prefetch(arow[rb] + kk + 128, 0, 3);   // WGP-scope, 4 k-steps ahead
        }
#pragma unroll
        for (int ct = 0; ct < 4; ++ct)
            b[ct] = *(const v16h*)(brow[ct] + kk);
#pragma unroll
        for (int rb = 0; rb < 4; ++rb)
#pragma unroll
            for (int ct = 0; ct < 4; ++ct)
                acc[rb][ct] = wmma16(a[rb], b[ct], acc[rb][ct]);
    }

    // ---- epilogue: wave-uniform routing
    const int which = f0 >> 10;                 // 0=q 1=k 2=v (uniform per wave)
    const int h     = (f0 >> 6) & 15;           // uniform per wave
    const int bb    = m0 >> 11;                 // uniform per block
    const int nbase = (m0 & (SEQ - 1));
    const size_t bh = (size_t)(bb * NHEADS + h);

    if (which == 0) {
        half_t* dst = Qh + bh * SEQ * HDIM;
#pragma unroll
        for (int ct = 0; ct < 4; ++ct) {
            int d = ct * 16 + ln;
#pragma unroll
            for (int rb = 0; rb < 4; ++rb)
#pragma unroll
                for (int r = 0; r < 8; ++r) {
                    int n = nbase + rb * 16 + r + half16 * 8;
                    dst[(size_t)n * HDIM + d] = (half_t)(acc[rb][ct][r] * SCALE);
                }
        }
    } else if (which == 1) {
        half_t* dst = Kh + bh * SEQ * HDIM;
#pragma unroll
        for (int ct = 0; ct < 4; ++ct) {
            int d = ct * 16 + ln;
#pragma unroll
            for (int rb = 0; rb < 4; ++rb)
#pragma unroll
                for (int r = 0; r < 8; ++r) {
                    int n = nbase + rb * 16 + r + half16 * 8;
                    dst[(size_t)n * HDIM + d] = (half_t)acc[rb][ct][r];
                }
        }
    } else {
        half_t* dst = Vth + bh * HDIM * SEQ;
#pragma unroll
        for (int ct = 0; ct < 4; ++ct) {
            int d = ct * 16 + ln;
#pragma unroll
            for (int rb = 0; rb < 4; ++rb)
#pragma unroll
                for (int r = 0; r < 8; ++r) {
                    int n = nbase + rb * 16 + r + half16 * 8;
                    dst[(size_t)d * SEQ + n] = (half_t)acc[rb][ct][r];
                }
        }
    }
}

// ---------------------------------------------------------------- flash attention
__global__ __launch_bounds__(128) void attn_kernel(const half_t* __restrict__ Qh,
                                                   const half_t* __restrict__ Kh,
                                                   const half_t* __restrict__ Vth,
                                                   half_t* __restrict__ attnh) {
    __shared__ alignas(32) half_t Plds[4][16][64];   // per-wave P tile

    const int lane   = threadIdx.x & 31;
    const int wv     = threadIdx.x >> 5;
    const int half16 = lane >> 4;
    const int ln     = lane & 15;
    const int bh     = blockIdx.x >> 5;              // SEQ/64 = 32 q-tiles
    const int qt     = blockIdx.x & 31;
    const int q0     = qt * 64 + wv * 16;

    const half_t* Qb = Qh  + (size_t)bh * SEQ * HDIM;
    const half_t* Kb = Kh  + (size_t)bh * SEQ * HDIM;
    const half_t* Vb = Vth + (size_t)bh * HDIM * SEQ;

    const half_t* qrow = Qb + (size_t)(q0 + ln) * HDIM;
    v16h qa0 = load_a_frag(qrow, 0, half16);
    v16h qa1 = load_a_frag(qrow, 32, half16);

    float mi[8], li[8];
    v8f zero = {};
    v8f o[4];
#pragma unroll
    for (int r = 0; r < 8; ++r) { mi[r] = -INFINITY; li[r] = 0.f; }
#pragma unroll
    for (int t = 0; t < 4; ++t) o[t] = zero;

    for (int k0 = 0; k0 < SEQ; k0 += 64) {
        // ---- S = Q @ K^T (16x64 per wave)
        v8f s[4];
#pragma unroll
        for (int t = 0; t < 4; ++t) s[t] = zero;
#pragma unroll
        for (int t = 0; t < 4; ++t) {
            const half_t* krow = Kb + (size_t)(k0 + t * 16 + ln) * HDIM + half16 * 16;
            v16h b0 = *(const v16h*)(krow);
            v16h b1 = *(const v16h*)(krow + 32);
            s[t] = wmma16(qa0, b0, s[t]);
            s[t] = wmma16(qa1, b1, s[t]);
        }

        // ---- online softmax per row (rows replicated across the 16 lanes of a half)
#pragma unroll
        for (int r = 0; r < 8; ++r) {
            float v0 = fmaxf(fmaxf(s[0][r], s[1][r]), fmaxf(s[2][r], s[3][r]));
#pragma unroll
            for (int off = 8; off >= 1; off >>= 1)
                v0 = fmaxf(v0, __shfl_xor(v0, off, 32));
            float mnew = fmaxf(mi[r], v0);
            float corr = __expf(mi[r] - mnew);
            mi[r] = mnew;
            float p0 = __expf(s[0][r] - mnew);
            float p1 = __expf(s[1][r] - mnew);
            float p2 = __expf(s[2][r] - mnew);
            float p3 = __expf(s[3][r] - mnew);
            s[0][r] = p0; s[1][r] = p1; s[2][r] = p2; s[3][r] = p3;
            float rs = (p0 + p1) + (p2 + p3);
#pragma unroll
            for (int off = 8; off >= 1; off >>= 1)
                rs += __shfl_xor(rs, off, 32);
            li[r] = li[r] * corr + rs;
            o[0][r] = o[0][r] * corr;
            o[1][r] = o[1][r] * corr;
            o[2][r] = o[2][r] * corr;
            o[3][r] = o[3][r] * corr;
        }

        // ---- P: C-layout -> A-layout via LDS
#pragma unroll
        for (int t = 0; t < 4; ++t)
#pragma unroll
            for (int r = 0; r < 8; ++r)
                Plds[wv][r + half16 * 8][t * 16 + ln] = (half_t)s[t][r];
        asm volatile("s_wait_dscnt 0" ::: "memory");

        // ---- O += P @ V  (Vt rows contiguous over keys)
        const half_t* prow = &Plds[wv][ln][0];
#pragma unroll
        for (int kh = 0; kh < 2; ++kh) {
            v16h pa = load_a_frag(prow, kh * 32, half16);
#pragma unroll
            for (int t = 0; t < 4; ++t) {
                const half_t* vrow = Vb + (size_t)(t * 16 + ln) * SEQ
                                        + k0 + kh * 32 + half16 * 16;
                v16h vb = *(const v16h*)vrow;
                o[t] = wmma16(pa, vb, o[t]);
            }
        }
    }

    // ---- finalize, store to [B*N, DIM] with head-contiguous columns
    const int b = bh >> 4, h = bh & 15;
#pragma unroll
    for (int r = 0; r < 8; ++r) {
        float inv = 1.0f / li[r];
        int n = q0 + r + half16 * 8;
        size_t row = ((size_t)b * SEQ + n) * DIM + h * HDIM;
#pragma unroll
        for (int t = 0; t < 4; ++t)
            attnh[row + t * 16 + ln] = (half_t)(o[t][r] * inv);
    }
}

// ---------------------------------------------------------------- proj GEMM + bias
// 64(M) x 64(N) per wave, fp32 output with bias.
__global__ __launch_bounds__(128) void proj_kernel(const half_t* __restrict__ ah,
                                                   const half_t* __restrict__ wh,
                                                   const float* __restrict__ bias,
                                                   float* __restrict__ out) {
    const int lane   = threadIdx.x & 31;
    const int wv     = threadIdx.x >> 5;
    const int half16 = lane >> 4;
    const int ln     = lane & 15;
    const int m0     = blockIdx.x * 64;
    const int f0     = blockIdx.y * 256 + wv * 64;

    const half_t* arow[4];
#pragma unroll
    for (int rb = 0; rb < 4; ++rb)
        arow[rb] = ah + (size_t)(m0 + rb * 16 + ln) * DIM;
    const half_t* brow[4];
#pragma unroll
    for (int ct = 0; ct < 4; ++ct)
        brow[ct] = wh + (size_t)(f0 + ct * 16 + ln) * DIM + half16 * 16;

    v8f zero = {};
    v8f acc[4][4];
#pragma unroll
    for (int rb = 0; rb < 4; ++rb)
#pragma unroll
        for (int ct = 0; ct < 4; ++ct) acc[rb][ct] = zero;

    for (int kk = 0; kk < DIM; kk += 32) {
        v16h a[4], b[4];
#pragma unroll
        for (int rb = 0; rb < 4; ++rb) {
            a[rb] = load_a_frag(arow[rb], kk, half16);
            __builtin_prefetch(arow[rb] + kk + 128, 0, 3);
        }
#pragma unroll
        for (int ct = 0; ct < 4; ++ct)
            b[ct] = *(const v16h*)(brow[ct] + kk);
#pragma unroll
        for (int rb = 0; rb < 4; ++rb)
#pragma unroll
            for (int ct = 0; ct < 4; ++ct)
                acc[rb][ct] = wmma16(a[rb], b[ct], acc[rb][ct]);
    }

#pragma unroll
    for (int ct = 0; ct < 4; ++ct) {
        int f = f0 + ct * 16 + ln;
        float bf = bias[f];
#pragma unroll
        for (int rb = 0; rb < 4; ++rb)
#pragma unroll
            for (int r = 0; r < 8; ++r) {
                int m = m0 + rb * 16 + r + half16 * 8;
                out[(size_t)m * DIM + f] = acc[rb][ct][r] + bf;
            }
    }
}

// ---------------------------------------------------------------- launch
extern "C" void kernel_launch(void* const* d_in, const int* in_sizes, int n_in,
                              void* d_out, int out_size, void* d_ws, size_t ws_size,
                              hipStream_t stream) {
    const float* x      = (const float*)d_in[0];
    const float* w_qkv  = (const float*)d_in[1];
    const float* w_proj = (const float*)d_in[2];
    const float* b_proj = (const float*)d_in[3];
    float* out = (float*)d_out;

    // workspace carve-up (f16 halfs)
    half_t* xh     = (half_t*)d_ws;
    half_t* wqkvh  = xh     + (size_t)TOK * DIM;        // 4.19M
    half_t* wprojh = wqkvh  + (size_t)F3 * DIM;         // 3.15M
    half_t* Qh     = wprojh + (size_t)DIM * DIM;        // 1.05M
    half_t* Kh     = Qh     + (size_t)TOK * DIM;        // Q/K/Vt each 4.19M
    half_t* Vth    = Kh     + (size_t)TOK * DIM;
    half_t* attnh  = Vth    + (size_t)TOK * DIM;        // total ~48 MB

    const int nx = TOK * DIM;
    const int nq = F3 * DIM;
    const int np = DIM * DIM;
    f32_to_f16_kernel<<<(nx + 255) / 256, 256, 0, stream>>>(x, xh, nx);
    f32_to_f16_kernel<<<(nq + 255) / 256, 256, 0, stream>>>(w_qkv, wqkvh, nq);
    f32_to_f16_kernel<<<(np + 255) / 256, 256, 0, stream>>>(w_proj, wprojh, np);

    qkv_kernel<<<dim3(TOK / 64, F3 / 256), 128, 0, stream>>>(xh, wqkvh, Qh, Kh, Vth);
    attn_kernel<<<BATCH * NHEADS * (SEQ / 64), 128, 0, stream>>>(Qh, Kh, Vth, attnh);
    proj_kernel<<<dim3(TOK / 64, DIM / 256), 128, 0, stream>>>(attnh, wprojh, b_proj, out);
}
